// TSPGAT_38714835206280
// MI455X (gfx1250) — matrix operand order repaired and checked by reading, not compiled, not run on hardware
//
#include <hip/hip_runtime.h>

typedef __attribute__((ext_vector_type(16))) __bf16 v16bf;
typedef __attribute__((ext_vector_type(8)))  float  v8f;

union Frag32 { uint4 u[2]; v16bf v; };

__device__ __forceinline__ unsigned short f2bf(float x) {
  unsigned u = __float_as_uint(x);
  unsigned r = (u + 0x7FFFu + ((u >> 16) & 1u)) >> 16;
  return (unsigned short)r;
}
__device__ __forceinline__ float bf2f(unsigned short h) {
  return __uint_as_float(((unsigned)h) << 16);
}

#define BM 128
#define BN 64
#define BK 32
#define LDK 40  // padded k-stride (ushorts); 80B rows keep 16B alignment

// C[M,Nout] = A[M,K] @ W[K,Nout], fp32 in/out, split-bf16 (hi/lo) WMMA core,
// double-buffered LDS pipeline. If gsrc != nullptr: A is the virtual gathered
// matrix A[r][k] = feat[(k<512 ? gsrc[r] : gdst[r])*512 + (k&511)] (edge head).
__global__ __launch_bounds__(256)
void gemm_split_bf16(const float* __restrict__ A, const float* __restrict__ W,
                     float* __restrict__ Out, int M, int K, int Nout,
                     const int* __restrict__ gsrc, const int* __restrict__ gdst,
                     const float* __restrict__ feat) {
  __shared__ __align__(16) unsigned short lA[2][2][BM][LDK];  // [buf][hi/lo][m][k]
  __shared__ __align__(16) unsigned short lB[2][2][BN][LDK];  // [buf][hi/lo][n][k]
  const int tid  = threadIdx.x;
  const int m0   = blockIdx.x * BM;
  const int n0   = blockIdx.y * BN;
  const int wid  = tid >> 5, lane = tid & 31;
  const int waveM = wid & 3, waveN = wid >> 2;
  const int half = lane >> 4, l16 = lane & 15;

  v8f acc[2][2];
#pragma unroll
  for (int mt = 0; mt < 2; ++mt)
#pragma unroll
    for (int nt = 0; nt < 2; ++nt)
#pragma unroll
      for (int i = 0; i < 8; ++i) acc[mt][nt][i] = 0.0f;

  float4 areg[4];  // staged A tile slice: 4 float4 / thread (128x32 tile)
  float4 breg[2];  // staged W tile slice: 2 float4 / thread (32x64 tile)

  // ---- b128 global staging into registers ----
  auto loadA = [&](int k0) {
    if (gsrc) {
#pragma unroll
      for (int i = 0; i < 4; ++i) {
        int idx4 = tid + i * 256;
        int row = idx4 >> 3, col4 = (idx4 & 7) * 4;
        int r = m0 + row, k = k0 + col4;  // whole BK chunk on one side of 512
        float4 v = make_float4(0.f, 0.f, 0.f, 0.f);
        if (r < M) {
          int node = (k < 512) ? gsrc[r] : gdst[r];
          v = *reinterpret_cast<const float4*>(feat + (size_t)node * 512 + (k & 511));
        }
        areg[i] = v;
      }
    } else {
#pragma unroll
      for (int i = 0; i < 4; ++i) {
        int idx4 = tid + i * 256;
        int row = idx4 >> 3, col4 = (idx4 & 7) * 4;
        int r = m0 + row;
        float4 v = make_float4(0.f, 0.f, 0.f, 0.f);
        if (r < M) v = *reinterpret_cast<const float4*>(A + (size_t)r * K + k0 + col4);
        areg[i] = v;
      }
    }
  };
  auto loadB = [&](int k0) {
#pragma unroll
    for (int i = 0; i < 2; ++i) {
      int idx4 = tid + i * 256;
      int kr = idx4 >> 4, nc4 = (idx4 & 15) * 4;
      breg[i] = *reinterpret_cast<const float4*>(W + (size_t)(k0 + kr) * Nout + n0 + nc4);
    }
  };
  // ---- fp32 -> bf16 hi/lo conversion + LDS store (A packed as b64) ----
  auto storeA = [&](int buf) {
#pragma unroll
    for (int i = 0; i < 4; ++i) {
      int idx4 = tid + i * 256;
      int row = idx4 >> 3, col4 = (idx4 & 7) * 4;
      float f[4] = {areg[i].x, areg[i].y, areg[i].z, areg[i].w};
      unsigned hi01, hi23, lo01, lo23;
      {
        unsigned short h0 = f2bf(f[0]), h1 = f2bf(f[1]), h2 = f2bf(f[2]), h3 = f2bf(f[3]);
        unsigned short l0 = f2bf(f[0] - bf2f(h0)), l1 = f2bf(f[1] - bf2f(h1));
        unsigned short l2 = f2bf(f[2] - bf2f(h2)), l3 = f2bf(f[3] - bf2f(h3));
        hi01 = (unsigned)h0 | ((unsigned)h1 << 16);
        hi23 = (unsigned)h2 | ((unsigned)h3 << 16);
        lo01 = (unsigned)l0 | ((unsigned)l1 << 16);
        lo23 = (unsigned)l2 | ((unsigned)l3 << 16);
      }
      uint2 uh; uh.x = hi01; uh.y = hi23;
      uint2 ul; ul.x = lo01; ul.y = lo23;
      *reinterpret_cast<uint2*>(&lA[buf][0][row][col4]) = uh;
      *reinterpret_cast<uint2*>(&lA[buf][1][row][col4]) = ul;
    }
  };
  auto storeB = [&](int buf) {
#pragma unroll
    for (int i = 0; i < 2; ++i) {
      int idx4 = tid + i * 256;
      int kr = idx4 >> 4, nc4 = (idx4 & 15) * 4;
      float f[4] = {breg[i].x, breg[i].y, breg[i].z, breg[i].w};
#pragma unroll
      for (int j = 0; j < 4; ++j) {  // transpose: scatter across n rows
        unsigned short h = f2bf(f[j]);
        lB[buf][0][nc4 + j][kr] = h;
        lB[buf][1][nc4 + j][kr] = f2bf(f[j] - bf2f(h));
      }
    }
  };

  // ---- software pipeline: prologue ----
  loadA(0);
  loadB(0);
  storeA(0);
  storeB(0);
  __syncthreads();

  int cur = 0;
  for (int k0 = 0; k0 < K; k0 += BK) {
    const bool more = (k0 + BK) < K;
    if (more) {  // issue next tile's global loads; they complete under the WMMAs
      loadA(k0 + BK);
      loadB(k0 + BK);
    }

    // ---- fragments from LDS (CDNA5 16-bit A and B: two 8-elem K runs) ----
    Frag32 aH[2], aL[2], bH[2], bL[2];
#pragma unroll
    for (int mt = 0; mt < 2; ++mt) {
      int mr = waveM * 32 + mt * 16 + l16;
      aH[mt].u[0] = *reinterpret_cast<const uint4*>(&lA[cur][0][mr][half * 8]);
      aH[mt].u[1] = *reinterpret_cast<const uint4*>(&lA[cur][0][mr][16 + half * 8]);
      aL[mt].u[0] = *reinterpret_cast<const uint4*>(&lA[cur][1][mr][half * 8]);
      aL[mt].u[1] = *reinterpret_cast<const uint4*>(&lA[cur][1][mr][16 + half * 8]);
    }
#pragma unroll
    for (int nt = 0; nt < 2; ++nt) {
      int nr = waveN * 32 + nt * 16 + l16;
      const uint4* p = reinterpret_cast<const uint4*>(&lB[cur][0][nr][half * 8]);
      bH[nt].u[0] = p[0]; bH[nt].u[1] = p[2];
      const uint4* q = reinterpret_cast<const uint4*>(&lB[cur][1][nr][half * 8]);
      bL[nt].u[0] = q[0]; bL[nt].u[1] = q[2];
    }

#pragma unroll
    for (int mt = 0; mt < 2; ++mt)
#pragma unroll
      for (int nt = 0; nt < 2; ++nt) {
        acc[mt][nt] = __builtin_amdgcn_wmma_f32_16x16x32_bf16(
            false, aH[mt].v, false, bH[nt].v, (short)0, acc[mt][nt], false, false);
        acc[mt][nt] = __builtin_amdgcn_wmma_f32_16x16x32_bf16(
            false, aH[mt].v, false, bL[nt].v, (short)0, acc[mt][nt], false, false);
        acc[mt][nt] = __builtin_amdgcn_wmma_f32_16x16x32_bf16(
            false, aL[mt].v, false, bH[nt].v, (short)0, acc[mt][nt], false, false);
      }

    if (more) {  // fill the other buffer; barrier publishes it for next iter
      storeA(cur ^ 1);
      storeB(cur ^ 1);
      __syncthreads();
      cur ^= 1;
    }
  }

  // ---- epilogue: C layout lane n=l16, vgpr v -> m = v + half*8 ----
#pragma unroll
  for (int mt = 0; mt < 2; ++mt)
#pragma unroll
    for (int nt = 0; nt < 2; ++nt)
#pragma unroll
      for (int v = 0; v < 8; ++v) {
        int rm = m0 + waveM * 32 + mt * 16 + half * 8 + v;
        int cn = n0 + waveN * 32 + nt * 16 + l16;
        if (rm < M && cn < Nout) Out[(size_t)rm * Nout + cn] = acc[mt][nt][v];
      }
}

// ---------------- attention scores: s_src/s_dst [N,8] ----------------
__global__ void gat_scores(const float* __restrict__ h, const float* __restrict__ a,
                           float* __restrict__ ssrc, float* __restrict__ sdst, int N) {
  int t = blockIdx.x * blockDim.x + threadIdx.x;
  if (t >= N * 8) return;
  int n = t >> 3, hd = t & 7;
  const float4* hp = reinterpret_cast<const float4*>(h + (size_t)n * 512 + hd * 64);
  const float4* a1 = reinterpret_cast<const float4*>(a + hd * 128);
  const float4* a2 = reinterpret_cast<const float4*>(a + hd * 128 + 64);
  float s1 = 0.f, s2 = 0.f;
#pragma unroll
  for (int i = 0; i < 16; ++i) {
    float4 hv = hp[i], av = a1[i], bv = a2[i];
    s1 += hv.x * av.x + hv.y * av.y + hv.z * av.z + hv.w * av.w;
    s2 += hv.x * bv.x + hv.y * bv.y + hv.z * bv.z + hv.w * bv.w;
  }
  ssrc[t] = s1;
  sdst[t] = s2;
}

// ---------------- CSR build (in-edges by dst, self-loops appended) ----------------
__device__ __forceinline__ int edge_src(int e, const int* esrc, int E) { return e < E ? esrc[e] : e - E; }
__device__ __forceinline__ int edge_dst(int e, const int* edst, int E) { return e < E ? edst[e] : e - E; }

__global__ void zero_i32(int* p, int n) {
  int i = blockIdx.x * blockDim.x + threadIdx.x;
  if (i < n) p[i] = 0;
}

__global__ void csr_count(const int* __restrict__ edst, int E, int Etot, int* __restrict__ deg) {
  int e = blockIdx.x * blockDim.x + threadIdx.x;
  if (e < Etot) atomicAdd(&deg[edge_dst(e, edst, E)], 1);
}

__global__ void csr_scan(const int* __restrict__ deg, int* __restrict__ rowstart, int n) {
  __shared__ int sm[1024];
  int t = threadIdx.x;
  int per = (n + 1023) >> 10;
  int base = t * per;
  int s = 0;
  for (int j = 0; j < per; ++j) { int idx = base + j; if (idx < n) s += deg[idx]; }
  sm[t] = s;
  __syncthreads();
  for (int off = 1; off < 1024; off <<= 1) {
    int v = (t >= off) ? sm[t - off] : 0;
    __syncthreads();
    sm[t] += v;
    __syncthreads();
  }
  int run = (t == 0) ? 0 : sm[t - 1];
  for (int j = 0; j < per; ++j) {
    int idx = base + j;
    if (idx < n) { rowstart[idx] = run; run += deg[idx]; }
  }
  if (t == 1023) rowstart[n] = run;
}

__global__ void csr_scatter(const int* __restrict__ edst, int E, int Etot,
                            const int* __restrict__ rowstart, int* __restrict__ cursor,
                            int* __restrict__ eid) {
  int e = blockIdx.x * blockDim.x + threadIdx.x;
  if (e >= Etot) return;
  int d = edge_dst(e, edst, E);
  int pos = rowstart[d] + atomicAdd(&cursor[d], 1);
  eid[pos] = e;
}

// ---------------- per-node softmax over in-edges (one wave32 / node) ----------------
__global__ void gat_softmax(const int* __restrict__ rowstart, const int* __restrict__ eid,
                            const int* __restrict__ esrc, int E,
                            const float* __restrict__ ssrc, const float* __restrict__ sdst,
                            float* __restrict__ attn, int N) {
  int gw = (blockIdx.x * blockDim.x + threadIdx.x) >> 5;
  int lane = threadIdx.x & 31;
  if (gw >= N) return;
  int beg = rowstart[gw], end = rowstart[gw + 1];
  float sd[8];
  {
    const float4* p = reinterpret_cast<const float4*>(sdst + (size_t)gw * 8);
    float4 x = p[0], y = p[1];
    sd[0] = x.x; sd[1] = x.y; sd[2] = x.z; sd[3] = x.w;
    sd[4] = y.x; sd[5] = y.y; sd[6] = y.z; sd[7] = y.w;
  }
  float mx[8];
#pragma unroll
  for (int h = 0; h < 8; ++h) mx[h] = -3.4e38f;
  for (int i = beg + lane; i < end; i += 32) {
    int e = eid[i];
    int s = edge_src(e, esrc, E);
    const float4* p = reinterpret_cast<const float4*>(ssrc + (size_t)s * 8);
    float4 x = p[0], y = p[1];
    float ss[8] = {x.x, x.y, x.z, x.w, y.x, y.y, y.z, y.w};
#pragma unroll
    for (int h = 0; h < 8; ++h) {
      float v = ss[h] + sd[h];
      v = v > 0.f ? v : 0.2f * v;  // leaky_relu(0.2)
      mx[h] = fmaxf(mx[h], v);
    }
  }
#pragma unroll
  for (int off = 16; off; off >>= 1)
#pragma unroll
    for (int h = 0; h < 8; ++h) mx[h] = fmaxf(mx[h], __shfl_xor(mx[h], off, 32));

  float sum[8];
#pragma unroll
  for (int h = 0; h < 8; ++h) sum[h] = 0.f;
  for (int i = beg + lane; i < end; i += 32) {
    int e = eid[i];
    int s = edge_src(e, esrc, E);
    const float4* p = reinterpret_cast<const float4*>(ssrc + (size_t)s * 8);
    float4 x = p[0], y = p[1];
    float ss[8] = {x.x, x.y, x.z, x.w, y.x, y.y, y.z, y.w};
#pragma unroll
    for (int h = 0; h < 8; ++h) {
      float v = ss[h] + sd[h];
      v = v > 0.f ? v : 0.2f * v;
      float ex = expf(v - mx[h]);
      attn[(size_t)e * 8 + h] = ex;
      sum[h] += ex;
    }
  }
#pragma unroll
  for (int off = 16; off; off >>= 1)
#pragma unroll
    for (int h = 0; h < 8; ++h) sum[h] += __shfl_xor(sum[h], off, 32);
  float inv[8];
#pragma unroll
  for (int h = 0; h < 8; ++h) inv[h] = 1.0f / sum[h];
  for (int i = beg + lane; i < end; i += 32) {
    int e = eid[i];
#pragma unroll
    for (int h = 0; h < 8; ++h) attn[(size_t)e * 8 + h] *= inv[h];
  }
}

// ---------------- aggregation out[n] = sum_e attn[e,h]*h[src,h,:] (wave/node) ----------------
__global__ void gat_aggregate(const int* __restrict__ rowstart, const int* __restrict__ eid,
                              const int* __restrict__ esrc, int E,
                              const float* __restrict__ attn, const float* __restrict__ h,
                              float* __restrict__ out, int N) {
  int gw = (blockIdx.x * blockDim.x + threadIdx.x) >> 5;
  int lane = threadIdx.x & 31;
  if (gw >= N) return;
  int beg = rowstart[gw], end = rowstart[gw + 1];
  float acc[16];
#pragma unroll
  for (int k = 0; k < 16; ++k) acc[k] = 0.f;
  for (int i = beg; i < end; ++i) {
    int e = eid[i];
    int s = edge_src(e, esrc, E);
    float av = (lane < 8) ? attn[(size_t)e * 8 + lane] : 0.f;
    const float* hp = h + (size_t)s * 512;
#pragma unroll
    for (int k = 0; k < 16; ++k)  // channel c = lane + 32k  ->  head = k>>1 (uniform)
      acc[k] += __shfl(av, k >> 1, 32) * hp[lane + 32 * k];
  }
  float* op = out + (size_t)gw * 512;
#pragma unroll
  for (int k = 0; k < 16; ++k) op[lane + 32 * k] = acc[k];
}

// ---------------- BatchNorm (training-mode batch stats), deterministic 2-stage ----------------
__global__ void bn_stage1(const float* __restrict__ X, int R, int C, int RB,
                          float* __restrict__ psum, float* __restrict__ psq) {
  int c = blockIdx.x * 256 + threadIdx.x;
  if (c >= C) return;
  int r0 = blockIdx.y * RB;
  int r1 = r0 + RB; if (r1 > R) r1 = R;
  float s = 0.f, q = 0.f;
  for (int r = r0; r < r1; ++r) {
    float v = X[(size_t)r * C + c];
    s += v; q += v * v;
  }
  psum[(size_t)blockIdx.y * C + c] = s;
  psq [(size_t)blockIdx.y * C + c] = q;
}

__global__ void bn_stage2(const float* __restrict__ psum, const float* __restrict__ psq,
                          int C, int nb, float Rf,
                          const float* __restrict__ g, const float* __restrict__ b,
                          float* __restrict__ scale, float* __restrict__ shift) {
  int c = blockIdx.x * blockDim.x + threadIdx.x;
  if (c >= C) return;
  float s = 0.f, q = 0.f;
  for (int i = 0; i < nb; ++i) { s += psum[(size_t)i * C + c]; q += psq[(size_t)i * C + c]; }
  float mean = s / Rf;
  float var = q / Rf - mean * mean;
  float sc = g[c] * rsqrtf(var + 1e-5f);
  scale[c] = sc;
  shift[c] = b[c] - mean * sc;
}

// act: 0=none, 1=ELU, 2=ReLU
__global__ void bn_apply(const float* __restrict__ X, float* __restrict__ Y, size_t total, int C,
                         const float* __restrict__ scale, const float* __restrict__ shift, int act) {
  size_t i = (size_t)blockIdx.x * blockDim.x + threadIdx.x;
  if (i >= total) return;
  int c = (int)(i % (size_t)C);
  float v = X[i] * scale[c] + shift[c];
  if (act == 1) v = v > 0.f ? v : (expf(v) - 1.0f);
  else if (act == 2) v = fmaxf(v, 0.f);
  Y[i] = v;
}

__global__ void vec_add(const float* __restrict__ a, const float* __restrict__ b,
                        float* __restrict__ c, size_t n) {
  size_t i = (size_t)blockIdx.x * blockDim.x + threadIdx.x;
  if (i < n) c[i] = a[i] + b[i];
}

__global__ void edge_dot(const float* __restrict__ Mm, const float* __restrict__ W2,
                         float* __restrict__ v, int E) {
  int e = blockIdx.x * blockDim.x + threadIdx.x;
  if (e >= E) return;
  const float4* mp = reinterpret_cast<const float4*>(Mm + (size_t)e * 64);
  const float4* wp = reinterpret_cast<const float4*>(W2);
  float s = 0.f;
#pragma unroll
  for (int i = 0; i < 16; ++i) {
    float4 m = mp[i], w = wp[i];
    s += m.x * w.x + m.y * w.y + m.z * w.z + m.w * w.w;
  }
  v[e] = s;
}

extern "C" void kernel_launch(void* const* d_in, const int* in_sizes, int n_in,
                              void* d_out, int out_size, void* d_ws, size_t ws_size,
                              hipStream_t stream) {
  (void)in_sizes; (void)n_in; (void)out_size; (void)ws_size;
  const int N = 20000, E = 320000, DIN = 128, D = 512;
  const int Etot = E + N;

  const float* x = (const float*)d_in[0];
  const int* esrc = (const int*)d_in[1];       // edge_idx[0]
  const int* edst = esrc + E;                  // edge_idx[1]

  // ---- workspace carve-up ----
  char* ws = (char*)d_ws;
  size_t off = 0;
  auto alloc = [&](size_t bytes) -> char* {
    char* p = ws + off;
    off = (off + bytes + 255) & ~(size_t)255;
    return p;
  };
  float* bufA  = (float*)alloc((size_t)N * D * 4);  // block output / final h
  float* bufG  = (float*)alloc((size_t)N * D * 4);  // gemm out (pre-attention h)
  float* bufS  = (float*)alloc((size_t)N * D * 4);  // aggregation out / h1+h2
  float* bufH1 = (float*)alloc((size_t)N * D * 4);
  float* bufH2 = (float*)alloc((size_t)N * D * 4);
  float* ssrcB = (float*)alloc((size_t)N * 8 * 4);
  float* sdstB = (float*)alloc((size_t)N * 8 * 4);
  float* attn  = (float*)alloc((size_t)Etot * 8 * 4);
  int* degcur  = (int*)alloc((size_t)2 * N * 4);
  int* deg = degcur;
  int* cursor = degcur + N;
  int* rowstart = (int*)alloc((size_t)(N + 1) * 4);
  int* eid = (int*)alloc((size_t)Etot * 4);
  float* psum = (float*)alloc((size_t)160 * 512 * 4);
  float* psq  = (float*)alloc((size_t)160 * 512 * 4);
  float* scale = (float*)alloc(512 * 4);
  float* shift = (float*)alloc(512 * 4);
  // MLP phase aliases (bufG..bufS region = 2*40.96MB >= E*64*4; bufH1 free)
  float* mlpM = bufG;
  float* vOut = bufH1;

  // ---- CSR build over edges + self-loops (dst-grouped) ----
  zero_i32<<<(2 * N + 255) / 256, 256, 0, stream>>>(degcur, 2 * N);
  csr_count<<<(Etot + 255) / 256, 256, 0, stream>>>(edst, E, Etot, deg);
  csr_scan<<<1, 1024, 0, stream>>>(deg, rowstart, N);
  csr_scatter<<<(Etot + 255) / 256, 256, 0, stream>>>(edst, E, Etot, rowstart, cursor, eid);

  auto bn = [&](const float* X, float* Y, int R, int C, const float* g, const float* b, int act) {
    int RB = (C >= 512) ? 128 : 2048;
    int nb = (R + RB - 1) / RB;
    dim3 g1((C + 255) / 256, nb);
    bn_stage1<<<g1, 256, 0, stream>>>(X, R, C, RB, psum, psq);
    bn_stage2<<<(C + 63) / 64, 64, 0, stream>>>(psum, psq, C, nb, (float)R, g, b, scale, shift);
    size_t total = (size_t)R * C;
    bn_apply<<<(unsigned)((total + 255) / 256), 256, 0, stream>>>(X, Y, total, C, scale, shift, act);
  };

  auto gat = [&](const float* xin, int K, int pbase, float* outY) {
    const float* W = (const float*)d_in[pbase + 0];
    const float* a = (const float*)d_in[pbase + 1];
    const float* g = (const float*)d_in[pbase + 2];
    const float* b = (const float*)d_in[pbase + 3];
    dim3 gg((N + BM - 1) / BM, D / BN);
    gemm_split_bf16<<<gg, 256, 0, stream>>>(xin, W, bufG, N, K, D, nullptr, nullptr, nullptr);
    gat_scores<<<(N * 8 + 255) / 256, 256, 0, stream>>>(bufG, a, ssrcB, sdstB, N);
    gat_softmax<<<(N * 32 + 255) / 256, 256, 0, stream>>>(rowstart, eid, esrc, E, ssrcB, sdstB, attn, N);
    gat_aggregate<<<(N * 32 + 255) / 256, 256, 0, stream>>>(rowstart, eid, esrc, E, attn, bufG, bufS, N);
    bn(bufS, outY, N, D, g, b, /*ELU*/ 1);
  };

  // ---- 3 blocks of 3 GAT layers ----
  const float* cur = x;
  int K = DIN;
  for (int blk = 0; blk < 3; ++blk) {
    int pb = 2 + blk * 12;
    gat(cur, K, pb + 0, bufH1);                 // h1 = elu(bn(gat(x)))
    gat(bufH1, D, pb + 4, bufH2);               // h2 = elu(bn(gat(h1)))
    vec_add<<<(unsigned)(((size_t)N * D + 255) / 256), 256, 0, stream>>>(
        bufH1, bufH2, bufS, (size_t)N * D);     // h1 + h2
    gat(bufS, D, pb + 8, bufA);                 // elu(bn(gat(h1+h2)))
    cur = bufA;
    K = D;
  }

  // ---- edge MLP head over ORIGINAL edges ----
  const float* mW1 = (const float*)d_in[38];
  const float* mg1 = (const float*)d_in[39];
  const float* mb1 = (const float*)d_in[40];
  const float* mW2 = (const float*)d_in[41];
  const float* mg2 = (const float*)d_in[42];
  const float* mb2 = (const float*)d_in[43];

  dim3 ge((E + BM - 1) / BM, 64 / BN);
  gemm_split_bf16<<<ge, 256, 0, stream>>>(nullptr, mW1, mlpM, E, 2 * D, 64, esrc, edst, bufA);
  bn(mlpM, mlpM, E, 64, mg1, mb1, /*ReLU*/ 2);
  edge_dot<<<(E + 255) / 256, 256, 0, stream>>>(mlpM, mW2, vOut, E);
  bn(vOut, (float*)d_out, E, 1, mg2, mb2, /*none*/ 0);
}
